// GATEncoder_43542378447331
// MI455X (gfx1250) — matrix operand order repaired and checked by reading, not compiled, not run on hardware
//
#include <hip/hip_runtime.h>

// ---------------- problem constants (match reference) ----------------
#define NNODES 50000
#define GDIM   1000
#define NEDGE  800000
#define HEADS  8
#define CHID   16
#define HC     128       // HEADS*CHID
#define LATENT 16
#define NEGS   0.2f

typedef __attribute__((ext_vector_type(16))) __bf16        v16bf;
typedef __attribute__((ext_vector_type(8)))  float         v8f;
typedef __attribute__((ext_vector_type(4)))  unsigned int  v4u;
typedef __attribute__((ext_vector_type(8)))  int           v8i;
typedef __attribute__((ext_vector_type(4)))  int           v4i;

// =====================================================================
// TDM: issue a 2D tile load (32 x 128 elems, 4B each) into LDS.
// D# built per CDNA5 ISA 8.3/8.4.  Padding: 4 dwords after every 32
// dwords -> LDS row stride 36 dwords (144B, 16B aligned, bank-spread).
// OOB (beyond tensor_dim0/1) reads return zero -> handles K/M tails.
// =====================================================================
__device__ __forceinline__ void tdm_load_tile(const float* gaddr, unsigned lds_addr,
                                              int t_dim0, int t_dim1, int stride0)
{
    unsigned long long ga = (unsigned long long)(uintptr_t)gaddr;
    v4u g0;
    g0[0] = 1u;                                   // count=1, user mode
    g0[1] = lds_addr;                             // LDS byte address
    g0[2] = (unsigned)(ga & 0xFFFFFFFFull);       // global_addr[31:0]
    g0[3] = (unsigned)((ga >> 32) & 0x1FFFFFFull) // global_addr[56:32]
            | 0x80000000u;                        // type=2 ("image")
    v8i g1;
    g1[0] = (int)((2u << 16)      // data_size = 4B
                | (1u << 20)      // pad_enable
                | (4u << 22)      // pad_interval: every 32 dwords
                | (3u << 25));    // pad_amount: 4 dwords
    g1[1] = (int)(((unsigned)t_dim0 & 0xFFFFu) << 16);                    // tensor_dim0 lo
    g1[2] = (int)((((unsigned)t_dim0 >> 16) & 0xFFFFu)
                | (((unsigned)t_dim1 & 0xFFFFu) << 16));                  // dim0 hi | dim1 lo
    g1[3] = (int)((((unsigned)t_dim1 >> 16) & 0xFFFFu) | (32u << 16));    // dim1 hi | tile_dim0=32
    g1[4] = (int)(128u);                                                  // tile_dim1=128, tile_dim2=0
    g1[5] = stride0;                                                      // tensor_dim0_stride lo
    g1[6] = 0;
    g1[7] = 0;
    v4i g2 = {0, 0, 0, 0};
    v4i g3 = {0, 0, 0, 0};
    v8i g4 = {0, 0, 0, 0, 0, 0, 0, 0};
    __builtin_amdgcn_tensor_load_to_lds(g0, g1, g2, g3, g4, 0);
}

// =====================================================================
// W pre-convert: Wt[n][k] = bf16(W[k][n]), k zero-padded to Kpad.
// Layout is B-fragment native: each lane reads two 16B chunks, 32B apart, per K-step.
// =====================================================================
__global__ void conv_w_bf16(const float* __restrict__ W, __bf16* __restrict__ Wt,
                            int K, int Kpad, int Nc)
{
    int idx = blockIdx.x * blockDim.x + threadIdx.x;
    if (idx >= Nc * Kpad) return;
    int n = idx / Kpad, k = idx - n * Kpad;
    float w = (k < K) ? W[(size_t)k * Nc + n] : 0.0f;
    Wt[idx] = (__bf16)w;
}

// =====================================================================
// GEMM: C[M,Nc] = A[M,K](f32) x W[K,Nc], bf16 WMMA, f32 acc.
// 256 thr = 8 waves; tile 128 rows x BN cols; wave w owns rows w*16..+16.
// A tile via double-buffered TDM into LDS; B fragments from Wt with
// 4-wide load-group / wmma-group software pipelining.
// =====================================================================
template<int BN>
__global__ __launch_bounds__(256)
void gemm_wmma_tdm(const float* __restrict__ A, const __bf16* __restrict__ Wt,
                   float* __restrict__ C, int M, int K, int Kpad, int Nc)
{
    __shared__ float As[2][128 * 36];   // 2 x 18432B, padded rows (36 dw)

    const int tid  = threadIdx.x;
    const int wave = tid >> 5;
    const int lane = tid & 31;
    const int half = lane >> 4;
    const int ln16 = lane & 15;

    const int rowBase = blockIdx.x * 128;
    const int bn      = blockIdx.y * BN;
    const unsigned lds0 = (unsigned)(uintptr_t)&As[0][0];
    const int nk = (K + 31) >> 5;
    constexpr int NJ = BN / 16;

    if (wave == 0)
        tdm_load_tile(A + (size_t)rowBase * K, lds0, K, M - rowBase, K);

    v8f acc[NJ] = {};

    for (int i = 0; i < nk; ++i) {
        const int k0 = i << 5;
        if (wave == 0)
            __builtin_amdgcn_s_wait_tensorcnt(0);   // tile i landed
        __syncthreads();                            // publish to all waves
        if (wave == 0 && i + 1 < nk) {
            int k1 = k0 + 32;
            tdm_load_tile(A + (size_t)rowBase * K + k1,
                          lds0 + (unsigned)(((i + 1) & 1) * (128 * 36 * 4)),
                          K - k1, M - rowBase, K);
        }

        // ---- A fragment (branch-free): row = wave*16+ln16,
        //      half 0 -> K {0..7,16..23}, half 1 -> K {8..15,24..31}
        const float* rp = &As[i & 1][(wave * 16 + ln16) * 36 + half * 8];
        float4 q0 = *(const float4*)(rp);
        float4 q1 = *(const float4*)(rp + 4);
        float4 q2 = *(const float4*)(rp + 16);
        float4 q3 = *(const float4*)(rp + 20);
        float av[16] = { q0.x, q0.y, q0.z, q0.w, q1.x, q1.y, q1.z, q1.w,
                         q2.x, q2.y, q2.z, q2.w, q3.x, q3.y, q3.z, q3.w };
        v16bf afr;
#pragma unroll
        for (int t = 0; t < 16; ++t) afr[t] = (__bf16)av[t];

        // ---- B fragments from Wt: lane = column, half 0 -> K {0..7,16..23},
        //      half 1 -> K {8..15,24..31}.  Pipelined: 4 fragment loads (one
        //      clause), then 4 wmmas -> waits overlap earlier wmmas.
        const __bf16* wbase = Wt + (size_t)(bn + ln16) * Kpad + k0 + half * 8;
        union BB { uint4 q[2]; v16bf v; };
#pragma unroll
        for (int jj = 0; jj < NJ; jj += 4) {
            constexpr int GWMAX = (NJ < 4) ? NJ : 4;
            BB bgrp[GWMAX];
#pragma unroll
            for (int t = 0; t < GWMAX; ++t) {
                if (jj + t < NJ) {
                    const __bf16* wp = wbase + (size_t)(jj + t) * 16 * Kpad;
                    bgrp[t].q[0] = *(const uint4*)(wp);
                    bgrp[t].q[1] = *(const uint4*)(wp + 16);
                }
            }
#pragma unroll
            for (int t = 0; t < GWMAX; ++t) {
                if (jj + t < NJ) {
                    acc[jj + t] = __builtin_amdgcn_wmma_f32_16x16x32_bf16(
                        false, afr, false, bgrp[t].v, (short)0, acc[jj + t],
                        false, false);
                }
            }
        }
    }

    // ---- store: D layout: VGPR v -> row half*8+v, col = ln16
#pragma unroll
    for (int j = 0; j < NJ; ++j) {
#pragma unroll
        for (int v = 0; v < 8; ++v) {
            int r = rowBase + wave * 16 + half * 8 + v;
            if (r < M) C[(size_t)r * Nc + bn + j * 16 + ln16] = acc[j][v];
        }
    }
}

// =====================================================================
// Attention helpers (edge phase is L2-resident atomics)
// =====================================================================
__device__ __forceinline__ unsigned enc_f32(float f) {
    unsigned u = __float_as_uint(f);
    return (u & 0x80000000u) ? ~u : (u | 0x80000000u);
}
__device__ __forceinline__ float dec_f32(unsigned u) {
    return (u & 0x80000000u) ? __uint_as_float(u & 0x7FFFFFFFu)
                             : __uint_as_float(~u);
}

__global__ void fill_f32_kernel(float* __restrict__ p, float v, int n) {
    int i = blockIdx.x * blockDim.x + threadIdx.x;
    if (i < n) p[i] = v;
}
__global__ void fill_u32_kernel(unsigned* __restrict__ p, unsigned v, int n) {
    int i = blockIdx.x * blockDim.x + threadIdx.x;
    if (i < n) p[i] = v;
}

__global__ void node_attn_kernel(const float* __restrict__ h,
                                 const float* __restrict__ a_src,
                                 const float* __restrict__ a_dst,
                                 float* __restrict__ als, float* __restrict__ ald,
                                 int n, int Hh)
{
    int idx = blockIdx.x * blockDim.x + threadIdx.x;
    if (idx >= n * Hh) return;
    int node = idx / Hh, hh = idx - node * Hh;
    const float* hp  = h + (size_t)node * Hh * CHID + hh * CHID;
    const float* asw = a_src + hh * CHID;
    const float* adw = a_dst + hh * CHID;
    float s = 0.0f, d = 0.0f;
#pragma unroll
    for (int c = 0; c < CHID; ++c) { float v = hp[c]; s += v * asw[c]; d += v * adw[c]; }
    als[idx] = s; ald[idx] = d;
}

__global__ void edge_logits_max(const int* __restrict__ src, const int* __restrict__ dst,
                                const float* __restrict__ als, const float* __restrict__ ald,
                                float* __restrict__ logits, unsigned* __restrict__ maxb,
                                int e, int Hh)
{
    int idx = blockIdx.x * blockDim.x + threadIdx.x;
    if (idx >= e * Hh) return;
    int ee = idx / Hh, hh = idx - ee * Hh;
    int s = src[ee], d = dst[ee];
    float lg = als[s * Hh + hh] + ald[d * Hh + hh];
    lg = (lg > 0.0f) ? lg : NEGS * lg;
    logits[idx] = lg;
    atomicMax(&maxb[d * Hh + hh], enc_f32(lg));
}

__global__ void edge_exp_sum(const int* __restrict__ dst,
                             float* __restrict__ logits,
                             const unsigned* __restrict__ maxb,
                             float* __restrict__ denom, int e, int Hh)
{
    int idx = blockIdx.x * blockDim.x + threadIdx.x;
    if (idx >= e * Hh) return;
    int ee = idx / Hh, hh = idx - ee * Hh;
    int d = dst[ee];
    float m = dec_f32(maxb[d * Hh + hh]);
    float ev = __expf(logits[idx] - m);
    logits[idx] = ev;
    atomicAdd(&denom[d * Hh + hh], ev);
}

__global__ void edge_aggregate(const int* __restrict__ src, const int* __restrict__ dst,
                               const float* __restrict__ logits,
                               const float* __restrict__ denom,
                               const float* __restrict__ h,
                               float* __restrict__ agg, int e, int Hh)
{
    int idx = blockIdx.x * blockDim.x + threadIdx.x;
    if (idx >= e * Hh) return;
    int ee = idx / Hh, hh = idx - ee * Hh;
    int s = src[ee], d = dst[ee];
    float alpha = logits[idx] / (denom[d * Hh + hh] + 1e-16f);
    const float* hp = h + (size_t)s * Hh * CHID + hh * CHID;
    float* op = agg + (size_t)d * Hh * CHID + hh * CHID;
#pragma unroll
    for (int c = 0; c < CHID; ++c) atomicAdd(&op[c], alpha * hp[c]);
}

__global__ void bias_act_kernel(float* __restrict__ io, const float* __restrict__ b,
                                int n, int F, int do_elu)
{
    int idx = blockIdx.x * blockDim.x + threadIdx.x;
    if (idx >= n * F) return;
    float v = io[idx] + b[idx % F];
    if (do_elu) v = (v > 0.0f) ? v : (__expf(v) - 1.0f);
    io[idx] = v;
}

// =====================================================================
// Host orchestration
// =====================================================================
static inline int cdiv_i(long long a, long long b) { return (int)((a + b - 1) / b); }

static void run_attn_stage(hipStream_t stream, const float* h,
                           const float* a_s, const float* a_d, const float* b,
                           int Hh, const int* srcp, const int* dstp,
                           float* als, float* ald, unsigned* maxb, float* denom,
                           float* logits, float* agg, int do_elu)
{
    const int TB = 256;
    const int nh = NNODES * Hh;
    const int eh = NEDGE * Hh;
    const int nf = NNODES * Hh * CHID;

    node_attn_kernel<<<cdiv_i(nh, TB), TB, 0, stream>>>(h, a_s, a_d, als, ald, NNODES, Hh);
    fill_u32_kernel<<<cdiv_i(nh, TB), TB, 0, stream>>>(maxb, 0x007FFFFFu, nh); // enc(-inf)
    fill_f32_kernel<<<cdiv_i(nh, TB), TB, 0, stream>>>(denom, 0.0f, nh);
    fill_f32_kernel<<<cdiv_i(nf, TB), TB, 0, stream>>>(agg, 0.0f, nf);
    edge_logits_max<<<cdiv_i(eh, TB), TB, 0, stream>>>(srcp, dstp, als, ald, logits, maxb, NEDGE, Hh);
    edge_exp_sum<<<cdiv_i(eh, TB), TB, 0, stream>>>(dstp, logits, maxb, denom, NEDGE, Hh);
    edge_aggregate<<<cdiv_i(eh, TB), TB, 0, stream>>>(srcp, dstp, logits, denom, h, agg, NEDGE, Hh);
    bias_act_kernel<<<cdiv_i(nf, TB), TB, 0, stream>>>(agg, b, NNODES, Hh * CHID, do_elu);
}

extern "C" void kernel_launch(void* const* d_in, const int* in_sizes, int n_in,
                              void* d_out, int out_size, void* d_ws, size_t ws_size,
                              hipStream_t stream)
{
    (void)in_sizes; (void)n_in; (void)out_size; (void)ws_size;

    const float* x   = (const float*)d_in[0];
    const int*   ei  = (const int*)  d_in[1];
    const float* W0  = (const float*)d_in[2];
    const float* as0 = (const float*)d_in[3];
    const float* ad0 = (const float*)d_in[4];
    const float* b0  = (const float*)d_in[5];
    const float* W1  = (const float*)d_in[6];
    const float* as1 = (const float*)d_in[7];
    const float* ad1 = (const float*)d_in[8];
    const float* b1  = (const float*)d_in[9];
    const float* W2  = (const float*)d_in[10];
    const float* as2 = (const float*)d_in[11];
    const float* ad2 = (const float*)d_in[12];
    const float* b2  = (const float*)d_in[13];

    const int* srcp = ei;
    const int* dstp = ei + NEDGE;

    // ---- workspace carve-up (~84 MB) ----
    float*    gbuf   = (float*)d_ws;                         // GEMM result [N,128]
    float*    abuf   = gbuf + (size_t)NNODES * HC;           // aggregated  [N,128]
    float*    als    = abuf + (size_t)NNODES * HC;           // [N,H]
    float*    ald    = als + (size_t)NNODES * HEADS;
    unsigned* maxb   = (unsigned*)(ald + (size_t)NNODES * HEADS);
    float*    denom  = (float*)(maxb + (size_t)NNODES * HEADS);
    float*    logits = denom + (size_t)NNODES * HEADS;       // [E,H]
    __bf16*   wtb    = (__bf16*)(logits + (size_t)NEDGE * HEADS); // [Nc,Kpad] bf16

    const int TB = 256;
    const int KP0 = 1024;  // GDIM padded to 32

    // ---- layer 0: x[N,1000] @ W0 -> gbuf[N,128] ----
    conv_w_bf16<<<cdiv_i(HC * KP0, TB), TB, 0, stream>>>(W0, wtb, GDIM, KP0, HC);
    gemm_wmma_tdm<128><<<dim3(cdiv_i(NNODES, 128), 1), 256, 0, stream>>>(
        x, wtb, gbuf, NNODES, GDIM, KP0, HC);
    run_attn_stage(stream, gbuf, as0, ad0, b0, HEADS, srcp, dstp,
                   als, ald, maxb, denom, logits, abuf, 1);

    // ---- layer 1: abuf @ W1 -> gbuf ----
    conv_w_bf16<<<cdiv_i(HC * HC, TB), TB, 0, stream>>>(W1, wtb, HC, HC, HC);
    gemm_wmma_tdm<128><<<dim3(cdiv_i(NNODES, 128), 1), 256, 0, stream>>>(
        abuf, wtb, gbuf, NNODES, HC, HC, HC);
    run_attn_stage(stream, gbuf, as1, ad1, b1, HEADS, srcp, dstp,
                   als, ald, maxb, denom, logits, abuf, 1);

    // ---- layer 2: abuf @ W2 -> gbuf[N,16], attn (1 head) -> d_out ----
    conv_w_bf16<<<cdiv_i(LATENT * HC, TB), TB, 0, stream>>>(W2, wtb, HC, HC, LATENT);
    gemm_wmma_tdm<16><<<dim3(cdiv_i(NNODES, 128), 1), 256, 0, stream>>>(
        abuf, wtb, gbuf, NNODES, HC, HC, LATENT);
    run_attn_stage(stream, gbuf, as2, ad2, b2, 1, srcp, dstp,
                   als, ald, maxb, denom, logits, (float*)d_out, 0);
}
